// TCBlock_63831803953832
// MI455X (gfx1250) — compile-verified, matrix operand order checked by reading, not run on hardware
//
#include <hip/hip_runtime.h>

// ---------------------------------------------------------------------------
// FNO spectral conv for MI455X (gfx1250, wave32, WMMA bf16).
// Three bf16 WMMA GEMMs: forward 64-mode DFT, per-mode complex channel mix,
// inverse DFT fused with y + leaky_relu(y).
// ---------------------------------------------------------------------------

typedef __attribute__((ext_vector_type(16))) __bf16 v16bf;
typedef __attribute__((ext_vector_type(8)))  float  v8f;

union FragB {
    v16bf        v;
    unsigned int u[8];
    uint4        q[2];
};

#define WMMA_BF16(A, Bm, Cacc) \
    __builtin_amdgcn_wmma_f32_16x16x32_bf16(false, (A), false, (Bm), (short)0, (Cacc), false, false)

static constexpr int kB = 128;
static constexpr int kC = 512;
static constexpr int kM = 1024;
static constexpr int kModes = 64;
static constexpr int kRows = kB * kC;          // 65536
static constexpr float kTwoPiOver1024 = 6.13592315154256e-3f;

// workspace byte offsets
static constexpr size_t OFF_F = 0;                       // fwd DFT, swizzled B-frags: 256 KB
static constexpr size_t OFF_G = 262144;                  // inv DFT, swizzled B-frags: 256 KB
static constexpr size_t OFF_W = 524288;                  // weights bf16 swizzled:     64 MB
static constexpr size_t OFF_X = OFF_W + 67108864;        // X_ft bf16 [plane][b][c]:   16 MB
static constexpr size_t OFF_O = OFF_X + 16777216;        // O_ft bf16 [b][d][2i+ri]:   16 MB

__device__ __forceinline__ unsigned short f2bf(float f) {
    unsigned int u = __float_as_uint(f);
    u += 0x7FFFu + ((u >> 16) & 1u);          // round-to-nearest-even
    return (unsigned short)(u >> 16);
}
__device__ __forceinline__ unsigned int pack2(unsigned short lo, unsigned short hi) {
    return (unsigned int)lo | ((unsigned int)hi << 16);
}
// Even K offset (within a 32-wide K chunk) of bf16 pair held in A/B-frag dword j,
// for lane group g = lane/16.  (ISA 7.12.2, 16-bit A 16x32 layout.)
__device__ __forceinline__ int kpair_of(int j, int g) {
    return ((j & 3) << 1) + ((j >> 2) << 4) + (g << 3);
}

// ---------------------------------------------------------------------------
// Setup: forward DFT matrix F[m][2i+ri] (cos, -sin), pre-swizzled B-fragments.
// Index order: [coltile w:8][kchunk:32][lane:32][8 dwords].
__global__ void k_build_F(unsigned int* __restrict__ Fswz) {
    int t = blockIdx.x * 256 + threadIdx.x;   // 8192 threads
    int L = t & 31, kc = (t >> 5) & 31, w = t >> 10;
    int n = L & 15, g = L >> 4;
    int jg = w * 16 + n;
    int i = jg >> 1, reim = jg & 1;
    unsigned int* dst = Fswz + (size_t)t * 8;
#pragma unroll
    for (int e = 0; e < 8; ++e) {
        int kp = kpair_of(e, g);
        unsigned short h[2];
#pragma unroll
        for (int s = 0; s < 2; ++s) {
            int m = kc * 32 + kp + s;
            float th = (float)((i * m) & 1023) * kTwoPiOver1024;  // exact range reduction
            float val = reim ? -__sinf(th) : __cosf(th);
            h[s] = f2bf(val);
        }
        dst[e] = pack2(h[0], h[1]);
    }
}

// Setup: inverse DFT matrix G[2i+ri][m] with irfft weights a_i = (i==0?1:2)/M.
// Index order: [mtile:64][kchunk:4][lane:32][8 dwords].
__global__ void k_build_G(unsigned int* __restrict__ Gswz) {
    int t = blockIdx.x * 256 + threadIdx.x;   // 8192 threads
    int L = t & 31, kc = (t >> 5) & 3, mtile = t >> 7;
    int n = L & 15, g = L >> 4;
    int mcol = mtile * 16 + n;
    unsigned int* dst = Gswz + (size_t)t * 8;
#pragma unroll
    for (int e = 0; e < 8; ++e) {
        int kp = kpair_of(e, g);
        unsigned short h[2];
#pragma unroll
        for (int s = 0; s < 2; ++s) {
            int k = kc * 32 + kp + s;
            int i = k >> 1, reim = k & 1;
            float a = (i == 0 ? 1.0f : 2.0f) * (1.0f / 1024.0f);
            float th = (float)((i * mcol) & 1023) * kTwoPiOver1024;
            float val = reim ? -a * __sinf(th) : a * __cosf(th);
            h[s] = f2bf(val);
        }
        dst[e] = pack2(h[0], h[1]);
    }
}

// Setup: weight (C,C,modes,2) f32 -> bf16 swizzled B-fragments of W^T per mode.
// Index order: [mode:64][reim:2][dtile:32][kchunk:16][lane:32][8 dwords].
__global__ void k_build_W(const float* __restrict__ weight, unsigned int* __restrict__ Wswz) {
    int t = blockIdx.x * 256 + threadIdx.x;   // 2097152 threads
    int L = t & 31;
    int kc = (t >> 5) & 15;
    int dtile = (t >> 9) & 31;
    int reim = (t >> 14) & 1;
    int i = t >> 15;
    int n = L & 15, g = L >> 4;
    int d = dtile * 16 + n;
    unsigned int* dst = Wswz + (size_t)t * 8;
#pragma unroll
    for (int e = 0; e < 8; ++e) {
        int kp = kpair_of(e, g);
        unsigned short h[2];
#pragma unroll
        for (int s = 0; s < 2; ++s) {
            int c = kc * 32 + kp + s;
            float wv = weight[(((size_t)d * 512 + c) * 64 + i) * 2 + reim];
            h[s] = f2bf(wv);
        }
        dst[e] = pack2(h[0], h[1]);
    }
}

// ---------------------------------------------------------------------------
// K1: forward DFT.  (65536 x 1024) @ (1024 x 128) -> X_ft bf16 [2i+ri][b][c].
// Block = 8 waves: 16 rows x 128 cols; A-tile staged bf16 in LDS (32 KB).
__global__ void k_dft(const float* __restrict__ x,
                      const unsigned int* __restrict__ Fswz,
                      unsigned short* __restrict__ Xbf) {
    __shared__ unsigned int sA[16 * 512];     // 16 rows x 512 bf16-pairs = 32 KB
    const int tid = threadIdx.x;
    const int R0 = blockIdx.x * 16;

    const float2* xg = (const float2*)(x + (size_t)R0 * 1024);
#pragma unroll
    for (int it = 0; it < 32; ++it) {
        int slot = tid + it * 256;            // = row*512 + kpair
        float2 f = xg[slot];
        sA[slot] = pack2(f2bf(f.x), f2bf(f.y));
    }
    __syncthreads();

    const int L = tid & 31, w = tid >> 5;
    const int n = L & 15, g = L >> 4;
    v8f acc = {0.f, 0.f, 0.f, 0.f, 0.f, 0.f, 0.f, 0.f};

    for (int kc = 0; kc < 32; ++kc) {
        FragB a, bm;
        const int kb = kc * 16;
#pragma unroll
        for (int j = 0; j < 8; ++j)
            a.u[j] = sA[n * 512 + kb + (kpair_of(j, g) >> 1)];
        const uint4* bp = (const uint4*)(Fswz + (size_t)(((w * 32 + kc) * 32 + L)) * 8);
        bm.q[0] = bp[0];
        bm.q[1] = bp[1];
        acc = WMMA_BF16(a.v, bm.v, acc);
    }

    const int jg = w * 16 + n;                // plane index = 2*i + reim
#pragma unroll
    for (int r = 0; r < 8; ++r) {
        int row = R0 + r + 8 * g;
        int bI = row >> 9, cI = row & 511;
        Xbf[(size_t)jg * 65536 + (size_t)bI * 512 + cI] = f2bf(acc[r]);
    }
}

// ---------------------------------------------------------------------------
// K2: per-mode complex channel mix.  O_i = X_i @ W_i^T (4 real GEMMs fused).
// Block = one mode, 16 b-rows, 128 d-cols (8 waves).  W streams from L2.
__global__ void k_modemix(const unsigned int* __restrict__ Wswz,
                          const unsigned int* __restrict__ Xg,   // Xbf viewed as uint
                          unsigned int* __restrict__ Og) {
    __shared__ unsigned int sX[2][16][256];   // re/im x 16 b-rows x 256 c-pairs = 32 KB
    const int tid = threadIdx.x;
    const int bi = blockIdx.x;
    const int i = bi >> 5;
    const int rem = bi & 31;
    const int btile = rem >> 2, dgrp = rem & 3;
    const int b0 = btile * 16;

#pragma unroll
    for (int it = 0; it < 32; ++it) {
        int slot = tid + it * 256;            // 8192 uints
        int reim = slot >> 12;
        int rrow = (slot >> 8) & 15;
        int kp = slot & 255;
        sX[reim][rrow][kp] =
            Xg[(size_t)(2 * i + reim) * 32768 + (size_t)(b0 + rrow) * 256 + kp];
    }
    __syncthreads();

    const int L = tid & 31, w = tid >> 5;
    const int n = L & 15, g = L >> 4;
    const int dtile = dgrp * 8 + w;

    v8f aRR = {0.f,0.f,0.f,0.f,0.f,0.f,0.f,0.f};
    v8f aII = aRR, aRI = aRR, aIR = aRR;

    for (int kc = 0; kc < 16; ++kc) {
        FragB xre, xim, wre, wim;
        const int kb = kc * 16;
#pragma unroll
        for (int j = 0; j < 8; ++j) {
            int kp = kb + (kpair_of(j, g) >> 1);
            xre.u[j] = sX[0][n][kp];
            xim.u[j] = sX[1][n][kp];
        }
        size_t baseR = ((((size_t)(i * 2 + 0) * 32 + dtile) * 16 + kc) * 32 + L) * 8;
        size_t baseI = ((((size_t)(i * 2 + 1) * 32 + dtile) * 16 + kc) * 32 + L) * 8;
        const uint4* pr = (const uint4*)(Wswz + baseR);
        const uint4* pi = (const uint4*)(Wswz + baseI);
        wre.q[0] = pr[0]; wre.q[1] = pr[1];
        wim.q[0] = pi[0]; wim.q[1] = pi[1];
        __builtin_prefetch(Wswz + baseR + 256, 0, 1);   // next kc, global_prefetch_b8
        __builtin_prefetch(Wswz + baseI + 256, 0, 1);

        aRR = WMMA_BF16(xre.v, wre.v, aRR);
        aII = WMMA_BF16(xim.v, wim.v, aII);
        aRI = WMMA_BF16(xre.v, wim.v, aRI);
        aIR = WMMA_BF16(xim.v, wre.v, aIR);
    }

#pragma unroll
    for (int r = 0; r < 8; ++r) {
        int brow = b0 + r + 8 * g;
        int d = dtile * 16 + n;
        float ore = aRR[r] - aII[r];
        float oim = aRI[r] + aIR[r];
        // O layout: [b][d][2i+ri] bf16 -> one packed dword store per (re,im)
        Og[((size_t)brow * 512 + d) * 64 + i] = pack2(f2bf(ore), f2bf(oim));
    }
}

// ---------------------------------------------------------------------------
// K3: inverse DFT + fused epilogue.  (65536 x 128) @ (128 x 1024) -> out f32.
__global__ void k_idft(const unsigned int* __restrict__ Og,
                       const unsigned int* __restrict__ Gswz,
                       float* __restrict__ out) {
    __shared__ unsigned int sA[16 * 64];      // 16 rows x 64 pairs = 4 KB
    const int tid = threadIdx.x;
    const int rowtile = blockIdx.x >> 3;
    const int cg = blockIdx.x & 7;
    const int r0 = rowtile * 16;

#pragma unroll
    for (int it = 0; it < 4; ++it) {
        int slot = tid + it * 256;            // = row*64 + kpair
        sA[slot] = Og[(size_t)r0 * 64 + slot];
    }
    __syncthreads();

    const int L = tid & 31, w = tid >> 5;
    const int n = L & 15, g = L >> 4;
    const int mtile = cg * 8 + w;
    v8f acc = {0.f, 0.f, 0.f, 0.f, 0.f, 0.f, 0.f, 0.f};

#pragma unroll
    for (int kc = 0; kc < 4; ++kc) {
        FragB a, bm;
        const int kb = kc * 16;
#pragma unroll
        for (int j = 0; j < 8; ++j)
            a.u[j] = sA[n * 64 + kb + (kpair_of(j, g) >> 1)];
        const uint4* bp = (const uint4*)(Gswz + (size_t)((mtile * 4 + kc) * 32 + L) * 8);
        bm.q[0] = bp[0];
        bm.q[1] = bp[1];
        acc = WMMA_BF16(a.v, bm.v, acc);
    }

#pragma unroll
    for (int r = 0; r < 8; ++r) {
        int row = r0 + r + 8 * g;
        int mcol = mtile * 16 + n;
        float v = acc[r];
        // y + leaky_relu(y) == (y>0) ? 2y : 1.2y
        out[(size_t)row * 1024 + mcol] = v > 0.f ? 2.0f * v : 1.2f * v;
    }
}

// ---------------------------------------------------------------------------
extern "C" void kernel_launch(void* const* d_in, const int* in_sizes, int n_in,
                              void* d_out, int out_size, void* d_ws, size_t ws_size,
                              hipStream_t stream) {
    (void)in_sizes; (void)n_in; (void)out_size; (void)ws_size;
    const float* x      = (const float*)d_in[0];
    // d_in[1] = t_emb, unused by the reference math
    const float* weight = (const float*)d_in[2];
    float* out = (float*)d_out;

    char* ws = (char*)d_ws;
    unsigned int*   Fswz = (unsigned int*)(ws + OFF_F);
    unsigned int*   Gswz = (unsigned int*)(ws + OFF_G);
    unsigned int*   Wswz = (unsigned int*)(ws + OFF_W);
    unsigned short* Xbf  = (unsigned short*)(ws + OFF_X);
    unsigned int*   Og   = (unsigned int*)(ws + OFF_O);

    k_build_F<<<32, 256, 0, stream>>>(Fswz);
    k_build_G<<<32, 256, 0, stream>>>(Gswz);
    k_build_W<<<8192, 256, 0, stream>>>(weight, Wswz);

    k_dft<<<kRows / 16, 256, 0, stream>>>(x, Fswz, Xbf);                       // 4096 blocks
    k_modemix<<<kModes * 8 * 4, 256, 0, stream>>>(Wswz, (const unsigned int*)Xbf, Og); // 2048 blocks
    k_idft<<<(kRows / 16) * 8, 256, 0, stream>>>(Og, Gswz, out);               // 32768 blocks
}
